// ScoreNet_69372311765539
// MI455X (gfx1250) — matrix operand order repaired and checked by reading, not compiled, not run on hardware
//
#include <hip/hip_runtime.h>
#include <hip/hip_bf16.h>
#include <math.h>

typedef _Float16 v16h __attribute__((ext_vector_type(16)));
typedef _Float16 v8h  __attribute__((ext_vector_type(8)));
typedef _Float16 v4h  __attribute__((ext_vector_type(4)));
typedef float    v8f  __attribute__((ext_vector_type(8)));

#define BGRAPH 4096
#define NNODES 1048576
#define FEAT   44
#define TED    84
#define WAVES  8
#define NODES_PER_BLOCK (WAVES * 16)

__device__ __forceinline__ float swishf(float x) {
    // x * sigmoid(x) = x / (1 + e^-x)
    return x / (1.0f + __expf(-x));
}

// ---------------------------------------------------------------------------
// Pack a [NOUT, K] row-major f32 weight (torch convention, out x in) into the
// CDNA5 WMMA B-operand layout (16-bit B, 32xK16 tiles):
//   per (kt, nt) tile: lane l holds column n = 16*nt + (l & 15);
//   halves h=0..15 map to K = 32*kt + (h < 8 ? 0 : 16) + (l < 16 ? 0 : 8) + (h & 7).
// Stored as [(kt*NT + nt)][lane][16 halves] f16, so the main kernel can load
// one v16h (32B) per lane per WMMA.
// ---------------------------------------------------------------------------
__global__ void pack_weights(const float* __restrict__ src,
                             _Float16* __restrict__ dst,
                             int K, int NOUT, int NT, int KT) {
    int total = KT * NT * 32 * 16;
    int idx = blockIdx.x * blockDim.x + threadIdx.x;
    if (idx >= total) return;
    int h    = idx & 15;
    int lane = (idx >> 4) & 31;
    int nt   = (idx >> 9) % NT;
    int kt   = idx / (512 * NT);
    int col  = nt * 16 + (lane & 15);
    int k    = kt * 32 + ((h < 8) ? 0 : 16) + ((lane < 16) ? 0 : 8) + (h & 7);
    float v  = (col < NOUT) ? src[col * K + k] : 0.0f;
    dst[idx] = (_Float16)v;
}

// ---------------------------------------------------------------------------
// Per-graph time embedding: Fourier features -> Linear(84,84) -> swish, f16.
// Also computes inv_std[b] = 1/sqrt((sigma^(2t)-1)/(2 ln sigma)).
// ---------------------------------------------------------------------------
__global__ void embed_kernel(const float* __restrict__ t,
                             const float* __restrict__ W_f,
                             const float* __restrict__ ew,   // [TED,TED]
                             const float* __restrict__ eb,   // [TED]
                             _Float16* __restrict__ embed_h, // [B,TED]
                             float* __restrict__ inv_std) {  // [B]
    const int b   = blockIdx.x;
    const int tid = threadIdx.x;
    __shared__ float temb[TED];
    const float tv = t[b];
    const float TWO_PI = 6.28318530717958647692f;
    if (tid < TED) {
        float v;
        if (tid < TED / 2) v = __sinf(TWO_PI * tv * W_f[tid]);
        else               v = __cosf(TWO_PI * tv * W_f[tid - TED / 2]);
        temb[tid] = v;
    }
    if (tid == TED) {
        const float ln_s = 3.21887582486820075f;  // ln(25)
        float s2 = __expf(2.0f * tv * ln_s);
        inv_std[b] = rsqrtf((s2 - 1.0f) / (2.0f * ln_s));
    }
    __syncthreads();
    if (tid < TED) {
        float acc = eb[tid];
        const float* wr = ew + tid * TED;
        #pragma unroll 4
        for (int k = 0; k < TED; ++k) acc += temb[k] * wr[k];
        embed_h[(size_t)b * TED + tid] = (_Float16)swishf(acc);
    }
}

// ---------------------------------------------------------------------------
// Main fused MLP kernel. 8 waves/block, one 16-node tile per wave.
// Layer0: [16x128] x [128x64]  -> 16 v_wmma_f32_16x16x32_f16
// Layer1: [16x64]  x [64x32]   ->  4 WMMA
// Layer2: [16x32]  x [32x16p]  ->  1 WMMA (outputs padded 4->16)
// ---------------------------------------------------------------------------
__global__ __launch_bounds__(WAVES * 32)
void mlp_kernel(const float* __restrict__ node_attr,
                const int* __restrict__ gptr,          // [B+1]
                const _Float16* __restrict__ embed_h,  // [B,TED] f16
                const float* __restrict__ inv_std,     // [B]
                const _Float16* __restrict__ pw0,
                const _Float16* __restrict__ pw1,
                const _Float16* __restrict__ pw2,
                const float* __restrict__ b0,
                const float* __restrict__ b1,
                const float* __restrict__ b2,
                float* __restrict__ out) {
    __shared__ _Float16 sA [WAVES][16 * 128];
    __shared__ _Float16 sH1[WAVES][16 * 64];
    __shared__ _Float16 sH2[WAVES][16 * 32];
    __shared__ float    sInv[WAVES][16];

    const int tid     = threadIdx.x;
    const int wave    = tid >> 5;
    const int lane    = tid & 31;
    const int l16     = lane & 15;       // staging row / D-matrix column
    const int halfSel = lane >> 4;       // 0: cols 0..63, 1: cols 64..127
    const int kb      = halfSel * 8;     // A-operand K sub-offset per ISA layout
    const int rbase   = halfSel * 8;     // C/D rows e (+0) or e+8
    const int gbase   = (blockIdx.x * WAVES + wave) * 16;
    const int gn      = gbase + l16;

    // seg = searchsorted(ptr, n, 'right') - 1 : last i with ptr[i] <= n
    int lo = 0, hi = BGRAPH;
    while (lo < hi) {
        int mid = (lo + hi + 1) >> 1;
        if (gptr[mid] <= gn) lo = mid; else hi = mid - 1;
    }
    const int seg = lo;

    // ---- stage A = concat(node_attr f32->f16, embed_h f16) : [16 x 128] ----
    _Float16* sRow = &sA[wave][l16 * 128];
    const _Float16* em = embed_h + (size_t)seg * TED;
    if (halfSel == 0) {
        const float4* na4 = (const float4*)(node_attr + (size_t)gn * FEAT);
        #pragma unroll
        for (int i = 0; i < 11; ++i) {             // 44 floats, 16B loads
            float4 v = na4[i];
            v4h p;
            p[0] = (_Float16)v.x; p[1] = (_Float16)v.y;
            p[2] = (_Float16)v.z; p[3] = (_Float16)v.w;
            *(v4h*)(sRow + 4 * i) = p;             // 8B DS store
        }
        const uint2* e2 = (const uint2*)em;        // 8B-aligned (84*2 = 168)
        #pragma unroll
        for (int i = 0; i < 5; ++i)                // embed cols 0..19
            *(uint2*)(sRow + FEAT + 4 * i) = e2[i];
        sInv[wave][l16] = inv_std[seg];
    } else {
        const uint2* e2 = (const uint2*)(em + 20); // byte 40, 8B-aligned
        #pragma unroll
        for (int i = 0; i < 16; ++i)               // embed cols 20..83
            *(uint2*)(sRow + 64 + 4 * i) = e2[i];
    }
    __syncthreads();

    // ---- layer 0: A [16x128] f16, 4 K-tiles x 4 N-tiles ----
    // A-operand 16-bit layout: lane m=l16; lanes<16 halves = K{0..7,16..23},
    // lanes>=16 halves = K{8..15,24..31} (per K-tile of 32).
    const _Float16* aRow = &sA[wave][l16 * 128];
    v16h A0[4];
    #pragma unroll
    for (int kt = 0; kt < 4; ++kt) {
        v8h alo = *(const v8h*)(aRow + kt * 32 + kb);        // ds_load_b128
        v8h ahi = *(const v8h*)(aRow + kt * 32 + 16 + kb);   // ds_load_b128
        A0[kt] = __builtin_shufflevector(alo, ahi,
                 0, 1, 2, 3, 4, 5, 6, 7, 8, 9, 10, 11, 12, 13, 14, 15);
    }
    v8f acc0[4] = {};
    #pragma unroll
    for (int nt = 0; nt < 4; ++nt) {
        #pragma unroll
        for (int kt = 0; kt < 4; ++kt) {
            v16h B = *(const v16h*)(pw0 + (size_t)((kt * 4 + nt) * 32 + lane) * 16);
            acc0[nt] = __builtin_amdgcn_wmma_f32_16x16x32_f16(
                false, A0[kt], false, B, (short)0, acc0[nt], false, false);
        }
    }
    // epilogue: bias + swish, f32->f16 into sH1 [16 x 64] row-major
    #pragma unroll
    for (int nt = 0; nt < 4; ++nt) {
        int col = nt * 16 + l16;
        float bias = b0[col];
        #pragma unroll
        for (int e = 0; e < 8; ++e) {
            float x = acc0[nt][e] + bias;
            sH1[wave][(rbase + e) * 64 + col] = (_Float16)swishf(x);
        }
    }
    __syncthreads();

    // ---- layer 1: A [16x64], 2 K-tiles x 2 N-tiles ----
    const _Float16* h1Row = &sH1[wave][l16 * 64];
    v16h A1[2];
    #pragma unroll
    for (int kt = 0; kt < 2; ++kt) {
        v8h alo = *(const v8h*)(h1Row + kt * 32 + kb);
        v8h ahi = *(const v8h*)(h1Row + kt * 32 + 16 + kb);
        A1[kt] = __builtin_shufflevector(alo, ahi,
                 0, 1, 2, 3, 4, 5, 6, 7, 8, 9, 10, 11, 12, 13, 14, 15);
    }
    v8f acc1[2] = {};
    #pragma unroll
    for (int nt = 0; nt < 2; ++nt) {
        #pragma unroll
        for (int kt = 0; kt < 2; ++kt) {
            v16h B = *(const v16h*)(pw1 + (size_t)((kt * 2 + nt) * 32 + lane) * 16);
            acc1[nt] = __builtin_amdgcn_wmma_f32_16x16x32_f16(
                false, A1[kt], false, B, (short)0, acc1[nt], false, false);
        }
    }
    #pragma unroll
    for (int nt = 0; nt < 2; ++nt) {
        int col = nt * 16 + l16;
        float bias = b1[col];
        #pragma unroll
        for (int e = 0; e < 8; ++e) {
            float x = acc1[nt][e] + bias;
            sH2[wave][(rbase + e) * 32 + col] = (_Float16)swishf(x);
        }
    }
    __syncthreads();

    // ---- layer 2: A [16x32], 1 WMMA, outputs 0..3 valid ----
    const _Float16* h2Row = &sH2[wave][l16 * 32];
    v8h alo = *(const v8h*)(h2Row + kb);
    v8h ahi = *(const v8h*)(h2Row + 16 + kb);
    v16h A2 = __builtin_shufflevector(alo, ahi,
              0, 1, 2, 3, 4, 5, 6, 7, 8, 9, 10, 11, 12, 13, 14, 15);
    v16h B2 = *(const v16h*)(pw2 + (size_t)lane * 16);
    v8f acc2 = {};
    acc2 = __builtin_amdgcn_wmma_f32_16x16x32_f16(
        false, A2, false, B2, (short)0, acc2, false, false);

    // final epilogue: bias + swish + per-graph 1/std scaling, write [N,4] f32
    if (l16 < 4) {
        float bias = b2[l16];
        #pragma unroll
        for (int e = 0; e < 8; ++e) {
            int r = rbase + e;
            float x = acc2[e] + bias;
            float s = swishf(x) * sInv[wave][r];
            out[(size_t)(gbase + r) * 4 + l16] = s;
        }
    }
}

// ---------------------------------------------------------------------------
// Workspace layout (bytes):
//   [0, 1376256)            embed_h   : 4096*84 f16
//   [1376256, 1392640)      inv_std   : 4096 f32
//   [1392640, 1409024)      pw0       : 4*4*32*16 f16 (32B-aligned)
//   [1409024, 1413120)      pw1       : 2*2*32*16 f16
//   [1413120, 1414144)      pw2       : 1*1*32*16 f16
// ---------------------------------------------------------------------------
extern "C" void kernel_launch(void* const* d_in, const int* in_sizes, int n_in,
                              void* d_out, int out_size, void* d_ws, size_t ws_size,
                              hipStream_t stream) {
    const float* node_attr = (const float*)d_in[0];
    const float* t         = (const float*)d_in[1];
    const int*   gptr      = (const int*)d_in[2];
    const float* W_f       = (const float*)d_in[3];
    const float* ew        = (const float*)d_in[4];
    const float* eb        = (const float*)d_in[5];
    const float* w0        = (const float*)d_in[6];
    const float* b0        = (const float*)d_in[7];
    const float* w1        = (const float*)d_in[8];
    const float* b1        = (const float*)d_in[9];
    const float* w2        = (const float*)d_in[10];
    const float* b2        = (const float*)d_in[11];
    float* out = (float*)d_out;

    char* ws = (char*)d_ws;
    _Float16* embed_h = (_Float16*)ws;
    float*    inv_std = (float*)(ws + 1376256);
    _Float16* pw0     = (_Float16*)(ws + 1392640);
    _Float16* pw1     = (_Float16*)(ws + 1409024);
    _Float16* pw2     = (_Float16*)(ws + 1413120);

    pack_weights<<<(4 * 4 * 512 + 255) / 256, 256, 0, stream>>>(w0, pw0, 128, 64, 4, 4);
    pack_weights<<<(2 * 2 * 512 + 255) / 256, 256, 0, stream>>>(w1, pw1, 64, 32, 2, 2);
    pack_weights<<<(1 * 1 * 512 + 255) / 256, 256, 0, stream>>>(w2, pw2, 32, 4, 1, 1);

    embed_kernel<<<BGRAPH, 96, 0, stream>>>(t, W_f, ew, eb, embed_h, inv_std);

    mlp_kernel<<<NNODES / NODES_PER_BLOCK, WAVES * 32, 0, stream>>>(
        node_attr, gptr, embed_h, inv_std, pw0, pw1, pw2, b0, b1, b2, out);
}
